// CR_15831249453464
// MI455X (gfx1250) — compile-verified, matrix operand order checked by reading, not run on hardware
//
#include <hip/hip_runtime.h>
#include <math.h>

typedef _Float16 h16 __attribute__((ext_vector_type(16)));
typedef float    f8  __attribute__((ext_vector_type(8)));

#define NUQ 100001
#define NCQ 2001
#define NEQ 50001
#define NKQ 1001
#define TTQ 50
#define BBQ 2048
#define DDQ 64

// ---------------- graph propagation kernels (bandwidth/atomic bound) ----------------

__global__ __launch_bounds__(256) void k_init(const float* __restrict__ uF,
                                              const float* __restrict__ iF,
                                              float* __restrict__ cur,
                                              float* __restrict__ acc, int n_item) {
    long i = (long)blockIdx.x * 256 + threadIdx.x;
    long tot = (long)(NUQ + n_item) * DDQ;
    if (i >= tot) return;
    long node = i >> 6; int d = (int)(i & 63);
    float v = (node < NUQ) ? uF[node * DDQ + d] : iF[(node - NUQ) * DDQ + d];
    cur[i] = v; acc[i] = v;
}

__global__ __launch_bounds__(256) void k_zero(float* __restrict__ p, long n) {
    long i = (long)blockIdx.x * 256 + threadIdx.x;
    if (i < n) p[i] = 0.f;
}

// one wave32 per edge: lane l handles dims l and l+32 (fully coalesced atomics)
__global__ __launch_bounds__(256) void k_scatter(const int* __restrict__ src,
                                                 const int* __restrict__ dst,
                                                 const float* __restrict__ vals,
                                                 const float* __restrict__ cur,
                                                 float* __restrict__ nxt, int nnz) {
    long gid = (long)blockIdx.x * 256 + threadIdx.x;
    int e = (int)(gid >> 5);
    int lane = (int)(gid & 31);
    if (e >= nnz) return;
    int s = src[e], dd = dst[e];
    float v = vals[e];
    const float* cp = cur + (long)dd * DDQ;
    float* np = nxt + (long)s * DDQ;
    atomicAdd(&np[lane], v * cp[lane]);
    atomicAdd(&np[lane + 32], v * cp[lane + 32]);
}

__global__ __launch_bounds__(256) void k_add(float* __restrict__ acc,
                                             const float* __restrict__ nxt, long n) {
    long i = (long)blockIdx.x * 256 + threadIdx.x;
    if (i < n) acc[i] += nxt[i];
}

// users_rep += acc/3 ; item_out = acc_item/3
__global__ __launch_bounds__(256) void k_extract(const float* __restrict__ acc,
                                                 float* __restrict__ urep,
                                                 float* __restrict__ item_out, int n_item) {
    long i = (long)blockIdx.x * 256 + threadIdx.x;
    long tot = (long)(NUQ + n_item) * DDQ;
    if (i >= tot) return;
    float v = acc[i] * (1.f / 3.f);
    long node = i >> 6;
    if (node < NUQ) urep[i] += v;
    else if (item_out) item_out[i - (long)NUQ * DDQ] = v;
}

// ---------------- fused attention kernel: WMMA QKV + last-row softmax ----------------
// one block per batch element, 4 waves; M is padded 50->64, each wave owns one 16-row tile.

__global__ __launch_bounds__(128) void k_attn(const float* __restrict__ e1,
                                              const float* __restrict__ pe,
                                              const int* __restrict__ seq,
                                              const int* __restrict__ mask,
                                              const float* __restrict__ wq, const float* __restrict__ bq,
                                              const float* __restrict__ wk, const float* __restrict__ bk,
                                              const float* __restrict__ wv, const float* __restrict__ bv,
                                              float* __restrict__ att) {
    __shared__ _Float16 xh[64 * 72];      // x tile (padded rows = 0), f16 A operand
    __shared__ _Float16 wT[64 * 72];      // transposed weight tile (f16 B operand)
    __shared__ float qkv[3 * 64 * 65];    // q | k | v  (avoid array-of-LDS-pointers)
    __shared__ float lg[64];

    float* qs = qkv;
    float* ks = qkv + 64 * 65;
    float* vs = qkv + 2 * 64 * 65;

    int b = blockIdx.x;
    int tid = threadIdx.x;
    int lane = tid & 31, wave = tid >> 5;

    // x = e1[seq] + pe  (rows >= T are zero)
    for (int i = tid; i < 64 * 64; i += 128) {
        int t = i >> 6, d = i & 63;
        float v = 0.f;
        if (t < TTQ) {
            int e = seq[b * TTQ + t];
            v = e1[(long)e * DDQ + d] + pe[t * DDQ + d];
        }
        xh[t * 72 + d] = (_Float16)v;
    }

    const float* W[3]  = {wq, wk, wv};
    const float* Bs[3] = {bq, bk, bv};

    int m0 = wave * 16;
    int mrow = m0 + (lane & 15);
    int hi = lane >> 4;

    for (int wi = 0; wi < 3; ++wi) {
        float* Optr = qkv + wi * (64 * 65);   // runtime LDS pointer arithmetic
        __syncthreads();   // previous wT consumers done (also covers xh init)
        for (int i = tid; i < 64 * 64; i += 128) {
            int kk = i >> 6, n = i & 63;
            wT[n * 72 + kk] = (_Float16)W[wi][i];   // store transposed: wT[n][k]
        }
        __syncthreads();

        // A operand: 16-bit 16x32 layout — lane m=mrow, halves j: K = hi*8+j (j<8), 16+hi*8+j-8 (j>=8)
        h16 a0, a1;
        #pragma unroll
        for (int j = 0; j < 8; ++j) {
            a0[j]     = xh[mrow * 72 + hi * 8 + j];
            a0[j + 8] = xh[mrow * 72 + 16 + hi * 8 + j];
            a1[j]     = xh[mrow * 72 + 32 + hi * 8 + j];
            a1[j + 8] = xh[mrow * 72 + 48 + hi * 8 + j];
        }
        #pragma unroll
        for (int n0 = 0; n0 < 64; n0 += 16) {
            int ncol = n0 + (lane & 15);
            h16 b0, b1;
            #pragma unroll
            for (int j = 0; j < 8; ++j) {
                b0[j]     = wT[ncol * 72 + hi * 8 + j];
                b0[j + 8] = wT[ncol * 72 + 16 + hi * 8 + j];
                b1[j]     = wT[ncol * 72 + 32 + hi * 8 + j];
                b1[j + 8] = wT[ncol * 72 + 48 + hi * 8 + j];
            }
            f8 c = {};
            c = __builtin_amdgcn_wmma_f32_16x16x32_f16(false, a0, false, b0, (short)0, c, false, false);
            c = __builtin_amdgcn_wmma_f32_16x16x32_f16(false, a1, false, b1, (short)0, c, false, false);
            float bias = Bs[wi][ncol];
            #pragma unroll
            for (int r = 0; r < 8; ++r)   // D layout: N = lane&15, M = r + 8*hi
                Optr[(m0 + r + 8 * hi) * 65 + ncol] = c[r] + bias;
        }
    }
    __syncthreads();

    // masking (reference semantics: q,k rows with mask==0 -> -1e5)
    for (int i = tid; i < TTQ * 64; i += 128) {
        int t = i >> 6, d = i & 63;
        if (mask[b * TTQ + t] == 0) { qs[t * 65 + d] = -1e5f; ks[t * 65 + d] = -1e5f; }
    }
    __syncthreads();

    // only row t = T-1 of the attention output survives
    if (tid < TTQ) {
        float acc = 0.f;
        for (int d = 0; d < 64; ++d) acc += qs[(TTQ - 1) * 65 + d] * ks[tid * 65 + d];
        lg[tid] = acc * 0.125f;   // 1/sqrt(64)
    }
    __syncthreads();
    if (tid < 64) {
        float m = -1e30f;
        for (int s = 0; s < TTQ; ++s) m = fmaxf(m, lg[s]);
        float den = 0.f, o = 0.f;
        for (int s = 0; s < TTQ; ++s) {
            float p = __expf(lg[s] - m);
            den += p;
            o += p * vs[s * 65 + tid];
        }
        att[(long)b * DDQ + tid] = o / den;
    }
}

// ---------------- fused MLP (WMMA, K=128) + final per-course dot ----------------

__global__ __launch_bounds__(128) void k_mlp(const float* __restrict__ urep,
                                             const float* __restrict__ att,
                                             const float* __restrict__ c1,
                                             const float* __restrict__ mlp_w,
                                             const float* __restrict__ mlp_b,
                                             const int* __restrict__ user_ids,
                                             const int* __restrict__ course_ids,
                                             float* __restrict__ out) {
    __shared__ _Float16 uh[64 * 136];  // concat([u, att]) tile, 64 rows x 128 cols
    __shared__ _Float16 mT[64 * 136];  // mlp_w transposed: mT[n][k], k<128
    __shared__ float uf[64 * 65];

    int tid = threadIdx.x;
    int lane = tid & 31, wave = tid >> 5;
    int b0 = blockIdx.x * 64;

    for (int i = tid; i < 64 * 128; i += 128) {
        int r = i >> 7, kk = i & 127;
        int b = b0 + r;
        float v = (kk < 64) ? urep[(long)user_ids[b] * DDQ + kk]
                            : att[(long)b * DDQ + (kk - 64)];
        uh[r * 136 + kk] = (_Float16)v;
    }
    for (int i = tid; i < 128 * 64; i += 128) {
        int kk = i >> 6, n = i & 63;
        mT[n * 136 + kk] = (_Float16)mlp_w[i];
    }
    __syncthreads();

    int m0 = wave * 16;
    int mrow = m0 + (lane & 15);
    int hi = lane >> 4;

    h16 a[4];
    #pragma unroll
    for (int kb = 0; kb < 4; ++kb) {
        #pragma unroll
        for (int j = 0; j < 8; ++j) {
            a[kb][j]     = uh[mrow * 136 + kb * 32 + hi * 8 + j];
            a[kb][j + 8] = uh[mrow * 136 + kb * 32 + 16 + hi * 8 + j];
        }
    }
    #pragma unroll
    for (int n0 = 0; n0 < 64; n0 += 16) {
        int ncol = n0 + (lane & 15);
        f8 c = {};
        #pragma unroll
        for (int kb = 0; kb < 4; ++kb) {
            h16 bv;
            #pragma unroll
            for (int j = 0; j < 8; ++j) {
                bv[j]     = mT[ncol * 136 + kb * 32 + hi * 8 + j];
                bv[j + 8] = mT[ncol * 136 + kb * 32 + 16 + hi * 8 + j];
            }
            c = __builtin_amdgcn_wmma_f32_16x16x32_f16(false, a[kb], false, bv, (short)0, c, false, false);
        }
        float bias = mlp_b[ncol];
        #pragma unroll
        for (int r = 0; r < 8; ++r)
            uf[(m0 + r + 8 * hi) * 65 + ncol] = fmaxf(0.f, c[r] + bias);
    }
    __syncthreads();

    {   // out[b, ci] = dot(u_final[b], c1[course_ids[b, ci]])
        int r = tid >> 1, ci = tid & 1;
        int b = b0 + r;
        int cid = course_ids[b * 2 + ci];
        float acc = 0.f;
        for (int d = 0; d < 64; ++d) acc += uf[r * 65 + d] * c1[(long)cid * DDQ + d];
        out[b * 2 + ci] = acc;
    }
}

// ---------------- host orchestration ----------------

extern "C" void kernel_launch(void* const* d_in, const int* in_sizes, int n_in,
                              void* d_out, int out_size, void* d_ws, size_t ws_size,
                              hipStream_t stream) {
    const float* users_f  = (const float*)d_in[0];
    const float* courses_f= (const float*)d_in[1];
    const float* exer_f   = (const float*)d_in[2];
    const float* know_f   = (const float*)d_in[3];
    const float* wq_w = (const float*)d_in[4];
    const float* wq_b = (const float*)d_in[5];
    const float* wk_w = (const float*)d_in[6];
    const float* wk_b = (const float*)d_in[7];
    const float* wv_w = (const float*)d_in[8];
    const float* wv_b = (const float*)d_in[9];
    const float* mlp_w = (const float*)d_in[10];
    const float* mlp_b = (const float*)d_in[11];
    const float* pe    = (const float*)d_in[12];
    const float* uc_vals = (const float*)d_in[13];
    const float* ue_vals = (const float*)d_in[14];
    const float* uk_vals = (const float*)d_in[15];
    const int* uc_src = (const int*)d_in[16];
    const int* uc_dst = (const int*)d_in[17];
    const int* ue_src = (const int*)d_in[18];
    const int* ue_dst = (const int*)d_in[19];
    const int* uk_src = (const int*)d_in[20];
    const int* uk_dst = (const int*)d_in[21];
    const int* mask    = (const int*)d_in[22];
    const int* seq_idx = (const int*)d_in[23];
    const int* user_ids   = (const int*)d_in[24];
    const int* course_ids = (const int*)d_in[25];

    float* ws = (float*)d_ws;
    const long S = (long)(NUQ + NEQ) * DDQ;       // largest graph node count
    float* cur  = ws;
    float* nxt  = ws + S;
    float* acc  = ws + 2 * S;
    float* urep = ws + 3 * S;                     // NU*64
    float* e1   = urep + (long)NUQ * DDQ;         // NE*64
    float* c1   = e1 + (long)NEQ * DDQ;           // NC*64
    float* att  = c1 + (long)NCQ * DDQ;           // B*64

    // users_rep accumulated across the three graphs
    {
        long n = (long)NUQ * DDQ;
        k_zero<<<(unsigned)((n + 255) / 256), 256, 0, stream>>>(urep, n);
    }

    struct G { const float* itemF; int n_item; const int* src; const int* dst;
               const float* vals; int nnz; float* item_out; };
    G gs[3] = {
        { courses_f, NCQ, uc_src, uc_dst, uc_vals, in_sizes[13], c1 },
        { exer_f,    NEQ, ue_src, ue_dst, ue_vals, in_sizes[14], e1 },
        { know_f,    NKQ, uk_src, uk_dst, uk_vals, in_sizes[15], nullptr },
    };

    for (int g = 0; g < 3; ++g) {
        long tot = (long)(NUQ + gs[g].n_item) * DDQ;
        unsigned nb = (unsigned)((tot + 255) / 256);
        k_init<<<nb, 256, 0, stream>>>(users_f, gs[g].itemF, cur, acc, gs[g].n_item);
        float* c = cur; float* n = nxt;
        for (int layer = 0; layer < 2; ++layer) {
            k_zero<<<nb, 256, 0, stream>>>(n, tot);
            long thr = (long)gs[g].nnz * 32;
            k_scatter<<<(unsigned)((thr + 255) / 256), 256, 0, stream>>>(
                gs[g].src, gs[g].dst, gs[g].vals, c, n, gs[g].nnz);
            k_add<<<nb, 256, 0, stream>>>(acc, n, tot);
            float* t = c; c = n; n = t;
        }
        k_extract<<<nb, 256, 0, stream>>>(acc, urep, gs[g].item_out, gs[g].n_item);
    }

    k_attn<<<BBQ, 128, 0, stream>>>(e1, pe, seq_idx, mask,
                                    wq_w, wq_b, wk_w, wk_b, wv_w, wv_b, att);

    k_mlp<<<BBQ / 64, 128, 0, stream>>>(urep, att, c1, mlp_w, mlp_b,
                                        user_ids, course_ids, (float*)d_out);
}
